// TropicalAttention_12335146074812
// MI455X (gfx1250) — matrix operand order repaired and checked by reading, not compiled, not run on hardware
//
#include <hip/hip_runtime.h>
#include <hip/hip_bf16.h>

// Tropical attention, MI455X (gfx1250) version.
// logsumexp_j(W_ij + x_j) == log((exp(W) @ exp(x))_i)  -> all four tropical
// linears become f16 WMMA GEMMs with exp/log prologue/epilogue, with per-row
// max subtraction for numerical stability.

typedef _Float16 v16h __attribute__((ext_vector_type(16)));
typedef _Float16 v8h  __attribute__((ext_vector_type(8)));
typedef float    v8f  __attribute__((ext_vector_type(8)));

#define E_DIM   256
#define S_DIM   64
#define B_DIM   32
#define H_DIM   4
#define D_HEAD  64
#define NTOK    (B_DIM * S_DIM)   // 2048

// ---------------------------------------------------------------------------
// Stage 1: EW = (f16) exp(W), elementwise.
__global__ void trop_expw_f16(const float* __restrict__ W,
                              _Float16* __restrict__ EW, int n) {
  int i = blockIdx.x * blockDim.x + threadIdx.x;
  if (i < n) EW[i] = (_Float16)__expf(W[i]);
}

// ---------------------------------------------------------------------------
// Stage 2: per token row (256 wide): m = max_j x[j]; EX = (f16) exp(x - m).
// One wave32 per row; 8 rows per 256-thread block.
__global__ void trop_row_max_exp(const float* __restrict__ X,
                                 float* __restrict__ rowmax,
                                 _Float16* __restrict__ EX) {
  const int lane = threadIdx.x & 31;
  const int row  = blockIdx.x * 8 + (threadIdx.x >> 5);
  const float* xr = X + row * E_DIM;
  float v[8];
  float m = -1e30f;
#pragma unroll
  for (int u = 0; u < 8; ++u) {
    v[u] = xr[lane + 32 * u];
    m = fmaxf(m, v[u]);
  }
#pragma unroll
  for (int off = 16; off >= 1; off >>= 1)
    m = fmaxf(m, __shfl_xor(m, off, 32));
  _Float16* er = EX + row * E_DIM;
#pragma unroll
  for (int u = 0; u < 8; ++u)
    er[lane + 32 * u] = (_Float16)__expf(v[u] - m);
  if (lane == 0) rowmax[row] = m;
}

// ---------------------------------------------------------------------------
// Stage 3: Y[t,i] = log( sum_j EW[i,j] * EX[t,j] ) + rowmax[t] + bias[i].
// GEMM: (2048 x 256) x (256 x 256)^T via v_wmma_f32_16x16x32_f16.
// grid = (128 token tiles, 2 col clusters); 8 waves/block, 1 16x16 tile/wave.
__global__ void trop_lse_gemm(const _Float16* __restrict__ EX,  // NTOK x E
                              const _Float16* __restrict__ EW,  // E x E [i][j]
                              const float* __restrict__ rowmax,
                              const float* __restrict__ bias,
                              float* __restrict__ Y) {
  const int lane = threadIdx.x & 31;
  const int wave = threadIdx.x >> 5;
  const int t0 = blockIdx.x * 16;
  const int n0 = blockIdx.y * 128 + wave * 16;

  // Fragment maps in contiguous k order (wave32, 16-bit A 16x32 / B 32x16):
  const int mA  = lane & 15;
  const int kbA = (lane >> 4) * 16;   // A: lanes 16-31 hold K+16..K+31
  const int nB  = lane & 15;
  const int kbB = (lane >> 4) * 16;   // B: lanes 16-31 hold K+16..K+31

  const _Float16* arow = EX + (size_t)(t0 + mA) * E_DIM;
  const _Float16* brow = EW + (size_t)(n0 + nB) * E_DIM;  // B[k][n] = EW[n][k]

  v8f acc = {};
#pragma unroll
  for (int ks = 0; ks < E_DIM; ks += 32) {
    v16h a = *(const v16h*)(arow + ks + kbA);
    v16h b = *(const v16h*)(brow + ks + kbB);
    acc = __builtin_amdgcn_wmma_f32_16x16x32_f16(
        false, a, false, b, (short)0, acc, false, false);
  }

  // D layout: n = lane&15, m = r + (lane>=16 ? 8 : 0)
  const int nc = n0 + nB;
  const float bn = bias[nc];
  const int mb = (lane >> 4) * 8;
#pragma unroll
  for (int r = 0; r < 8; ++r) {
    const int t = t0 + mb + r;
    Y[(size_t)t * E_DIM + nc] = __logf(acc[r]) + rowmax[t] + bn;
  }
}

// ---------------------------------------------------------------------------
// Stage 4: per (batch, head): attn = softmax(q+k over D), AO = attn @ Vmat
// where Vmat[d][e] = v[b, token=d, h*64+e].  64x64x64 matmul via WMMA,
// A fragments read from LDS (f16, padded stride).
#define A_STR 72
#define SC_STR 68
__global__ void trop_attention(const float* __restrict__ Q,
                               const float* __restrict__ K,
                               const float* __restrict__ V,
                               float* __restrict__ AO) {
  __shared__ float    sc[S_DIM * SC_STR];
  __shared__ _Float16 at[S_DIM * A_STR];

  const int bb = blockIdx.x >> 2;  // batch
  const int hh = blockIdx.x & 3;   // head
  const int tid = threadIdx.x;

  if (tid < S_DIM) {
    const int s = tid;
    const float* qr = Q + (size_t)(bb * S_DIM + s) * E_DIM + hh * D_HEAD;
    const float* kr = K + (size_t)(bb * S_DIM + s) * E_DIM + hh * D_HEAD;
    float mx = -1e30f;
    for (int d = 0; d < D_HEAD; ++d) {
      const float z = qr[d] + kr[d];
      sc[s * SC_STR + d] = z;
      mx = fmaxf(mx, z);
    }
    float sum = 0.f;
    for (int d = 0; d < D_HEAD; ++d) {
      const float e = __expf(sc[s * SC_STR + d] - mx);
      sc[s * SC_STR + d] = e;
      sum += e;
    }
    const float inv = 1.f / sum;
    for (int d = 0; d < D_HEAD; ++d)
      at[s * A_STR + d] = (_Float16)(sc[s * SC_STR + d] * inv);
  }
  __syncthreads();

  const int lane = tid & 31;
  const int wave = tid >> 5;
  const int mA  = lane & 15;
  const int kbA = (lane >> 4) * 8;
  const int nB  = lane & 15;
  const int kbB = (lane >> 4) * 16;

#pragma unroll
  for (int hlf = 0; hlf < 2; ++hlf) {
    const int tile = wave + hlf * 8;       // 16 tiles in a 4x4 grid
    const int tm = tile >> 2, tn = tile & 3;
    v8f acc = {};
#pragma unroll
    for (int ks = 0; ks < D_HEAD; ks += 32) {
      const _Float16* arow = at + (tm * 16 + mA) * A_STR + ks;
      v8h alo = *(const v8h*)(arow + kbA);
      v8h ahi = *(const v8h*)(arow + 16 + kbA);
      v16h a;
#pragma unroll
      for (int h = 0; h < 8; ++h) { a[h] = alo[h]; a[h + 8] = ahi[h]; }
      v16h b;
#pragma unroll
      for (int h = 0; h < 16; ++h) {
        const int k = ks + (h < 8 ? kbA + h : 16 + kbA + h - 8);  // token index inside this batch slice
        b[h] = (_Float16)V[(size_t)(bb * S_DIM + k) * E_DIM + hh * D_HEAD +
                           tn * 16 + nB];
      }
      acc = __builtin_amdgcn_wmma_f32_16x16x32_f16(
          false, a, false, b, (short)0, acc, false, false);
    }
    const int mb = (lane >> 4) * 8;
#pragma unroll
    for (int r = 0; r < 8; ++r) {
      AO[(size_t)(bb * S_DIM + tm * 16 + mb + r) * E_DIM + hh * D_HEAD +
         tn * 16 + nB] = acc[r];
    }
  }
}

// ---------------------------------------------------------------------------
extern "C" void kernel_launch(void* const* d_in, const int* in_sizes, int n_in,
                              void* d_out, int out_size, void* d_ws,
                              size_t ws_size, hipStream_t stream) {
  (void)in_sizes; (void)n_in; (void)out_size; (void)ws_size;

  const float* x  = (const float*)d_in[0];
  const float* Wq = (const float*)d_in[1];
  const float* bq = (const float*)d_in[2];
  const float* Wk = (const float*)d_in[3];
  const float* bk = (const float*)d_in[4];
  const float* Wv = (const float*)d_in[5];
  const float* bv = (const float*)d_in[6];
  const float* Wo = (const float*)d_in[7];
  const float* bo = (const float*)d_in[8];
  float* out = (float*)d_out;

  // Workspace carve-up (~10.6 MB total).
  _Float16* EWq = (_Float16*)d_ws;
  _Float16* EWk = EWq + E_DIM * E_DIM;
  _Float16* EWv = EWk + E_DIM * E_DIM;
  _Float16* EWo = EWv + E_DIM * E_DIM;
  _Float16* EX  = EWo + E_DIM * E_DIM;        // NTOK*E f16
  _Float16* AEX = EX  + (size_t)NTOK * E_DIM; // NTOK*E f16
  float* xm = (float*)(AEX + (size_t)NTOK * E_DIM);
  float* am = xm + NTOK;
  float* q  = am + NTOK;
  float* k  = q + (size_t)NTOK * E_DIM;
  float* v  = k + (size_t)NTOK * E_DIM;
  float* ao = v + (size_t)NTOK * E_DIM;

  const dim3 blk(256);
  const dim3 gW(E_DIM * E_DIM / 256);
  trop_expw_f16<<<gW, blk, 0, stream>>>(Wq, EWq, E_DIM * E_DIM);
  trop_expw_f16<<<gW, blk, 0, stream>>>(Wk, EWk, E_DIM * E_DIM);
  trop_expw_f16<<<gW, blk, 0, stream>>>(Wv, EWv, E_DIM * E_DIM);
  trop_expw_f16<<<gW, blk, 0, stream>>>(Wo, EWo, E_DIM * E_DIM);

  trop_row_max_exp<<<NTOK / 8, blk, 0, stream>>>(x, xm, EX);

  const dim3 g3(NTOK / 16, 2);
  trop_lse_gemm<<<g3, blk, 0, stream>>>(EX, EWq, xm, bq, q);
  trop_lse_gemm<<<g3, blk, 0, stream>>>(EX, EWk, xm, bk, k);
  trop_lse_gemm<<<g3, blk, 0, stream>>>(EX, EWv, xm, bv, v);

  trop_attention<<<B_DIM * H_DIM, blk, 0, stream>>>(q, k, v, ao);

  trop_row_max_exp<<<NTOK / 8, blk, 0, stream>>>(ao, am, AEX);
  trop_lse_gemm<<<g3, blk, 0, stream>>>(AEX, EWo, am, bo, out);
}
